// GeometricTransformerModule_56831007261097
// MI455X (gfx1250) — matrix operand order repaired and checked by reading, not compiled, not run on hardware
//
#include <hip/hip_runtime.h>

typedef __attribute__((ext_vector_type(16))) _Float16 v16h;
typedef __attribute__((ext_vector_type(8)))  _Float16 v8h;
typedef __attribute__((ext_vector_type(8)))  float    v8f;
typedef __attribute__((ext_vector_type(4)))  float    v4f;

#define HIDD 256
#define NPTS 320
#define NBATCH 2
#define NHEADS 4
#define FFND 512

// ---------------------------------------------------------------------------
// Pack a KxN f32 weight into WMMA-B f16 fragments.
// Fragment (kkIdx, nt) covers K rows [32*kkIdx, +32) and cols [16*nt, +16).
// Storage: P[((kkIdx*NT + nt)*32 + lane)*8 + v]  (each lane's 8 dwords contiguous)
// Lane<16: K = 32*kkIdx + 2v+(v>=4?8:0),+1 ; Lane>=16: K = 32*kkIdx + 8 + 2v+(v>=4?8:0),+1 ; col = nt*16 + (lane&15)
// ---------------------------------------------------------------------------
__global__ void k_pack(const float* __restrict__ W, unsigned int* __restrict__ P,
                       int Kc, int Nc) {
  long t = (long)blockIdx.x * blockDim.x + threadIdx.x;
  long total = (long)Kc * Nc / 2;
  if (t >= total) return;
  int NT = Nc >> 4;
  long frag = t >> 8;
  int within = (int)(t & 255);
  int lane = within >> 3, v = within & 7;
  int kkIdx = (int)(frag / NT), nt = (int)(frag % NT);
  int krow = kkIdx * 32 + ((lane < 16) ? 0 : 8) + 2 * v + ((v >= 4) ? 8 : 0);
  int col  = nt * 16 + (lane & 15);
  float f0 = W[(long)krow * Nc + col];
  float f1 = W[((long)krow + 1) * Nc + col];
  union { unsigned int u; _Float16 h[2]; } cv;
  cv.h[0] = (_Float16)f0; cv.h[1] = (_Float16)f1;
  P[t] = cv.u;
}

// ---------------------------------------------------------------------------
// Generic GEMM: C[M,N] = act(A[M,K] @ Wpk + bias). One wave per block,
// 16 rows x 64 cols per wave, K stepped by 32 via v_wmma_f32_16x16x32_f16.
// B-fragment loads for all 4 n-tiles are grouped ahead of the 4 WMMAs, and
// the next k-step's fragments are prefetched (global_prefetch_b8).
// ---------------------------------------------------------------------------
__global__ __launch_bounds__(32) void k_gemm(
    const float* __restrict__ A, const unsigned int* __restrict__ Bpk,
    const float* __restrict__ bias, float* __restrict__ C,
    int M, int Nc, int Kc, int relu) {
  int lane = threadIdx.x & 31;
  int r0 = blockIdx.x * 16;
  int c0 = blockIdx.y * 64;
  int NT = Nc >> 4;
  int mrow = lane & 15;
  int khalf = (lane < 16) ? 0 : 8;
  const float* arow = A + (long)(r0 + mrow) * Kc;

  v8f acc[4];
  for (int nt = 0; nt < 4; ++nt)
    for (int j = 0; j < 8; ++j) acc[nt][j] = 0.0f;

  int nKK = Kc >> 5;
  for (int kk = 0; kk < nKK; ++kk) {
    int kb = kk * 32;
    v16h a;
#pragma unroll
    for (int j = 0; j < 8; ++j) {
      a[j]     = (_Float16)arow[kb + khalf + j];
      a[8 + j] = (_Float16)arow[kb + 16 + khalf + j];
    }
    if (kk + 1 < nKK) { // cover load->wmma latency for the next step
#pragma unroll
      for (int nt = 0; nt < 4; ++nt)
        __builtin_prefetch(Bpk + ((long)((kk + 1) * NT + (c0 >> 4) + nt) * 32 + lane) * 8, 0, 1);
      __builtin_prefetch(arow + kb + 32, 0, 1);
    }
    v16h bfr[4];
#pragma unroll
    for (int nt = 0; nt < 4; ++nt)
      bfr[nt] = *(const v16h*)(Bpk + ((long)(kk * NT + (c0 >> 4) + nt) * 32 + lane) * 8);
#pragma unroll
    for (int nt = 0; nt < 4; ++nt)
      acc[nt] = __builtin_amdgcn_wmma_f32_16x16x32_f16(
          false, a, false, bfr[nt], (short)0, acc[nt], false, false);
  }
  int col0 = lane & 15;
  int rb = (lane < 16) ? 0 : 8;
  for (int nt = 0; nt < 4; ++nt) {
    int col = c0 + nt * 16 + col0;
    float bv = bias ? bias[col] : 0.0f;
#pragma unroll
    for (int j = 0; j < 8; ++j) {
      float v = acc[nt][j] + bv;
      if (relu) v = fmaxf(v, 0.0f);
      C[(long)(r0 + rb + j) * Nc + col] = v;
    }
  }
}

// ---------------------------------------------------------------------------
// 3-NN reference vectors per (b,n): refs[b,n,k,3] = p[nn_k] - p[n]
// ---------------------------------------------------------------------------
__global__ void k_knn(const float* __restrict__ pts, float* __restrict__ refs,
                      int Bb, int Nn) {
  int t = blockIdx.x * blockDim.x + threadIdx.x;
  if (t >= Bb * Nn) return;
  int b = t / Nn, n = t % Nn;
  const float* pn = pts + ((long)b * Nn + n) * 3;
  float px = pn[0], py = pn[1], pz = pn[2];
  float bd0 = 1e30f, bd1 = 1e30f, bd2 = 1e30f;
  int bi0 = 0, bi1 = 0, bi2 = 0;
  for (int m = 0; m < Nn; ++m) {
    if (m == n) continue;
    const float* pm = pts + ((long)b * Nn + m) * 3;
    float dx = pm[0] - px, dy = pm[1] - py, dz = pm[2] - pz;
    float d2 = dx * dx + dy * dy + dz * dz;
    if (d2 < bd0)      { bd2=bd1; bi2=bi1; bd1=bd0; bi1=bi0; bd0=d2; bi0=m; }
    else if (d2 < bd1) { bd2=bd1; bi2=bi1; bd1=d2; bi1=m; }
    else if (d2 < bd2) { bd2=d2; bi2=m; }
  }
  int bi[3] = {bi0, bi1, bi2};
  float* rf = refs + (long)t * 9;
  for (int k = 0; k < 3; ++k) {
    const float* pk_ = pts + ((long)b * Nn + bi[k]) * 3;
    rf[k*3+0] = pk_[0] - px; rf[k*3+1] = pk_[1] - py; rf[k*3+2] = pk_[2] - pz;
  }
}

// ---------------------------------------------------------------------------
// Fused geometric structure embedding:
//   rpe[b,n,m,:] = sinenc(d)@proj_d + max_k sinenc(a_k)@proj_a    (f16 output)
// Block = 128 threads (4 waves); 16 rows of flattened (b,n,m) per block;
// wave w handles output cols [64w, 64w+64).
// k-step-outer / variant-inner: each B fragment pair (proj_d, proj_a) is
// loaded once and feeds 4 WMMAs. feat rows padded to 264 halves (528 B,
// = 4 dwords mod 64 banks) -> conflict-free A-fragment ds_load_b128.
// ---------------------------------------------------------------------------
__global__ __launch_bounds__(128) void k_gse(
    const float* __restrict__ pts, const float* __restrict__ refs,
    const unsigned int* __restrict__ pkD, const unsigned int* __restrict__ pkA,
    _Float16* __restrict__ rpe, int Nn) {
  __shared__ float freq[128];
  __shared__ float xsv[16][4];
  __shared__ __align__(16) _Float16 feat[4][16][264];

  int t = threadIdx.x;
  freq[t & 127] = __expf(-(float)(t & 127) * 0.071955784f); // ln(1e4)/128
  long r0 = (long)blockIdx.x * 16;
  if (t < 16) {
    long r = r0 + t;
    int m = (int)(r % Nn);
    int n = (int)((r / Nn) % Nn);
    int b = (int)(r / ((long)Nn * Nn));
    const float* pn = pts + ((long)b * Nn + n) * 3;
    const float* pm = pts + ((long)b * Nn + m) * 3;
    float ax = pn[0] - pm[0], ay = pn[1] - pm[1], az = pn[2] - pm[2]; // diffs = p[n]-p[m]
    float dist = sqrtf(ax * ax + ay * ay + az * az);
    xsv[t][0] = dist * 5.0f; // / SIGMA_D
    const float* rf = refs + ((long)b * Nn + n) * 9;
    for (int k = 0; k < 3; ++k) {
      float rx = rf[k*3+0], ry = rf[k*3+1], rz = rf[k*3+2];
      float cx = ry * az - rz * ay;
      float cy = rz * ax - rx * az;
      float cz = rx * ay - ry * ax;
      float sv = sqrtf(cx * cx + cy * cy + cz * cz);
      float cv = rx * ax + ry * ay + rz * az;
      xsv[t][1 + k] = atan2f(sv, cv) * 3.8197186342f; // FACTOR_A
    }
  }
  __syncthreads();

  { // cooperative feature fill: 4 variants x 16 rows x 256 f16
    int row = t >> 3;
    int j0 = (t & 7) * 32;
    for (int v = 0; v < 4; ++v) {
      float x = xsv[row][v];
      for (int j = 0; j < 32; ++j) {
        int jj = j0 + j;
        float f = (jj < 128) ? __sinf(x * freq[jj]) : __cosf(x * freq[jj - 128]);
        feat[v][row][jj] = (_Float16)f;
      }
    }
  }
  __syncthreads();

  int w = t >> 5, lane = t & 31;
  int mrow = lane & 15, khalf = (lane < 16) ? 0 : 8;
  v8f acc[4][4]; // [variant][ntile]
#pragma unroll
  for (int v = 0; v < 4; ++v)
    for (int nt = 0; nt < 4; ++nt)
      for (int j = 0; j < 8; ++j) acc[v][nt][j] = 0.f;

  for (int kk = 0; kk < 8; ++kk) {
    int kb = kk * 32;
    v16h afr[4];
#pragma unroll
    for (int v = 0; v < 4; ++v) {
      v8h lo = *(const v8h*)&feat[v][mrow][kb + khalf];
      v8h hi = *(const v8h*)&feat[v][mrow][kb + 16 + khalf];
#pragma unroll
      for (int j = 0; j < 8; ++j) { afr[v][j] = lo[j]; afr[v][8 + j] = hi[j]; }
    }
#pragma unroll
    for (int nt = 0; nt < 4; ++nt) {
      int ntg = w * 4 + nt; // NT = 16
      long fo = ((long)(kk * 16 + ntg) * 32 + lane) * 8;
      v16h bD = *(const v16h*)(pkD + fo);
      v16h bA = *(const v16h*)(pkA + fo);
      acc[0][nt] = __builtin_amdgcn_wmma_f32_16x16x32_f16(false, afr[0], false, bD, (short)0, acc[0][nt], false, false);
      acc[1][nt] = __builtin_amdgcn_wmma_f32_16x16x32_f16(false, afr[1], false, bA, (short)0, acc[1][nt], false, false);
      acc[2][nt] = __builtin_amdgcn_wmma_f32_16x16x32_f16(false, afr[2], false, bA, (short)0, acc[2][nt], false, false);
      acc[3][nt] = __builtin_amdgcn_wmma_f32_16x16x32_f16(false, afr[3], false, bA, (short)0, acc[3][nt], false, false);
    }
  }

  int col0 = lane & 15, rb = (lane < 16) ? 0 : 8;
  for (int nt = 0; nt < 4; ++nt) {
    int col = w * 64 + nt * 16 + col0;
#pragma unroll
    for (int j = 0; j < 8; ++j) {
      float am = fmaxf(acc[1][nt][j], fmaxf(acc[2][nt][j], acc[3][nt][j]));
      rpe[(r0 + rb + j) * 256 + col] = (_Float16)(acc[0][nt][j] + am);
    }
  }
}

// ---------------------------------------------------------------------------
// Qr[b,n,h,c] = sum_d Q[b,n,h,d] * Wrpe[c, h*64+d]   (folds rpe-weight into Q)
// ---------------------------------------------------------------------------
__global__ __launch_bounds__(256) void k_qr(const float* __restrict__ Q,
                                            const float* __restrict__ W,
                                            float* __restrict__ Qr) {
  __shared__ float q[256];
  int row = blockIdx.x, c = threadIdx.x;
  q[c] = Q[(long)row * 256 + c];
  __syncthreads();
  for (int h = 0; h < 4; ++h) {
    float s = 0.f;
    const float* wr = W + (long)c * 256 + h * 64;
    const float* qh = q + h * 64;
    for (int d = 0; d < 64; ++d) s += qh[d] * wr[d];
    Qr[((long)row * 4 + h) * 256 + c] = s;
  }
}

// ---------------------------------------------------------------------------
// Fused scores (+optional rpe position term) + softmax. Block = (b,n),
// 320 threads = one per m, all 4 heads; rpe[b,n] row streamed once (b128),
// Qr block staged LDS-direct via GLOBAL_LOAD_ASYNC_TO_LDS_B128 (ASYNCcnt).
// ---------------------------------------------------------------------------
__global__ __launch_bounds__(320) void k_scores(
    const float* __restrict__ Q, const float* __restrict__ Kf,
    const float* __restrict__ Qr, const _Float16* __restrict__ rpe,
    float* __restrict__ attn, int Nn, int has_pos) {
  __shared__ float qs[4][64];
  __shared__ __align__(16) float qrs[4][256];
  __shared__ float buf[320];
  int n = blockIdx.x, b = blockIdx.y;
  int t = threadIdx.x;
  long row = (long)b * Nn + n;
  if (t < 256) qs[t >> 6][t & 63] = Q[row * 256 + t];
  if (has_pos && t < 256) {
    // async gather: 4 floats per lane, memory -> LDS, no VGPR round-trip
    unsigned lds_off = (unsigned)(unsigned long long)(&qrs[0][0] + t * 4);
    unsigned long long gaddr = (unsigned long long)(Qr + row * 1024 + t * 4);
    asm volatile("global_load_async_to_lds_b128 %0, %1, off"
                 :: "v"(lds_off), "v"(gaddr) : "memory");
  }
  asm volatile("s_wait_asynccnt 0x0" ::: "memory");
  __syncthreads();

  int m = t;
  const v4f* kf4 = (const v4f*)(Kf + ((long)b * Nn + m) * 256);
  float s[4];
  for (int h = 0; h < 4; ++h) {
    float c_ = 0.f;
    for (int dd = 0; dd < 16; ++dd) {
      v4f kv = kf4[h * 16 + dd];
#pragma unroll
      for (int j = 0; j < 4; ++j) c_ += qs[h][dd * 4 + j] * kv[j];
    }
    s[h] = c_;
  }
  if (has_pos) {
    const v8h* rp8 = (const v8h*)(rpe + (row * Nn + m) * 256);
    float p0 = 0.f, p1 = 0.f, p2 = 0.f, p3 = 0.f;
    for (int cc = 0; cc < 32; ++cc) {
      v8h hv = rp8[cc];
#pragma unroll
      for (int j = 0; j < 8; ++j) {
        float f = (float)hv[j];
        int c = cc * 8 + j;
        p0 += qrs[0][c] * f; p1 += qrs[1][c] * f;
        p2 += qrs[2][c] * f; p3 += qrs[3][c] * f;
      }
    }
    s[0] += p0; s[1] += p1; s[2] += p2; s[3] += p3;
  }
  for (int h = 0; h < 4; ++h) s[h] *= 0.125f; // 1/sqrt(64)

  for (int h = 0; h < 4; ++h) {
    buf[t] = s[h]; __syncthreads();
    for (int str = 256; str >= 1; str >>= 1) {
      if (t < str && t + str < 320) buf[t] = fmaxf(buf[t], buf[t + str]);
      __syncthreads();
    }
    float mx = buf[0]; __syncthreads();
    float e = __expf(s[h] - mx);
    buf[t] = e; __syncthreads();
    for (int str = 256; str >= 1; str >>= 1) {
      if (t < str && t + str < 320) buf[t] += buf[t + str];
      __syncthreads();
    }
    float sum = buf[0]; __syncthreads();
    attn[(((long)b * 4 + h) * Nn + n) * Nn + m] = e / sum;
  }
}

// out[b,n,h,d] = sum_m attn[b,h,n,m] * V[b,m,h,d]
__global__ __launch_bounds__(256) void k_av(const float* __restrict__ attn,
                                            const float* __restrict__ V,
                                            float* __restrict__ out, int Nn) {
  int n = blockIdx.x, b = blockIdx.y, t = threadIdx.x;
  int h = t >> 6;
  float s = 0.f;
  const float* at = attn + (((long)b * 4 + h) * Nn + n) * Nn;
  const float* vp = V + (long)b * Nn * 256 + t;
  for (int m = 0; m < Nn; ++m) s += at[m] * vp[(long)m * 256];
  out[((long)b * Nn + n) * 256 + t] = s;
}

// out = LN(x + y) * g + b   (out may alias x)
__global__ __launch_bounds__(256) void k_add_ln(
    const float* __restrict__ x, const float* __restrict__ y,
    const float* __restrict__ g, const float* __restrict__ bta,
    float* __restrict__ out) {
  __shared__ float red[256];
  int row = blockIdx.x, t = threadIdx.x;
  float v = x[(long)row * 256 + t] + y[(long)row * 256 + t];
  red[t] = v; __syncthreads();
  for (int str = 128; str >= 1; str >>= 1) { if (t < str) red[t] += red[t + str]; __syncthreads(); }
  float mean = red[0] * (1.0f / 256.0f); __syncthreads();
  float dv = v - mean;
  red[t] = dv * dv; __syncthreads();
  for (int str = 128; str >= 1; str >>= 1) { if (t < str) red[t] += red[t + str]; __syncthreads(); }
  float var = red[0] * (1.0f / 256.0f);
  float r = rsqrtf(var + 1e-5f);
  out[(long)row * 256 + t] = dv * r * g[t] + bta[t];
}

// ---------------------------------------------------------------------------
extern "C" void kernel_launch(void* const* d_in, const int* in_sizes, int n_in,
                              void* d_out, int out_size, void* d_ws, size_t ws_size,
                              hipStream_t stream) {
  (void)in_sizes; (void)n_in; (void)out_size; (void)ws_size;
  const int B = NBATCH, N = NPTS;
  const int rows = B * N; // 640

  // -------- input mapping (setup_inputs dict insertion order) --------
  int ii = 0;
  auto nxt = [&]() -> const float* { return (const float*)d_in[ii++]; };
  const float* src_feats  = nxt();
  const float* src_points = nxt();
  const float* tgt_feats  = nxt();
  const float* tgt_points = nxt();
  const float* inW = nxt();  const float* inB = nxt();
  const float* outW = nxt(); const float* outB = nxt();
  const float* projD = nxt(); const float* projA = nxt();
  struct SP { const float *qw,*qb,*kw,*kb,*vw,*vb,*rpew,*ow,*ob,*l1g,*l1b,*f1w,*f1b,*f2w,*f2b,*l2g,*l2b; } S[3];
  for (int i = 0; i < 3; ++i) {
    S[i].qw=nxt(); S[i].qb=nxt(); S[i].kw=nxt(); S[i].kb=nxt(); S[i].vw=nxt(); S[i].vb=nxt();
    S[i].rpew=nxt(); S[i].ow=nxt(); S[i].ob=nxt(); S[i].l1g=nxt(); S[i].l1b=nxt();
    S[i].f1w=nxt(); S[i].f1b=nxt(); S[i].f2w=nxt(); S[i].f2b=nxt(); S[i].l2g=nxt(); S[i].l2b=nxt();
  }
  struct CP { const float *qw,*qb,*kw,*kb,*vw,*vb,*ow,*ob,*lag,*lab,*f1w,*f1b,*f2w,*f2b,*lfg,*lfb; } Cp[3];
  for (int i = 0; i < 3; ++i) {
    Cp[i].qw=nxt(); Cp[i].qb=nxt(); Cp[i].kw=nxt(); Cp[i].kb=nxt(); Cp[i].vw=nxt(); Cp[i].vb=nxt();
    Cp[i].ow=nxt(); Cp[i].ob=nxt(); Cp[i].lag=nxt(); Cp[i].lab=nxt();
    Cp[i].f1w=nxt(); Cp[i].f1b=nxt(); Cp[i].f2w=nxt(); Cp[i].f2b=nxt(); Cp[i].lfg=nxt(); Cp[i].lfb=nxt();
  }

  // -------- workspace layout --------
  char* wsp = (char*)d_ws;
  size_t off = 0;
  auto alloc = [&](size_t bytes) -> void* {
    void* p = wsp + off;
    off = (off + bytes + 255) & ~(size_t)255;
    return p;
  };
  auto pku = [&](int K, int Nc) { return (unsigned int*)alloc((size_t)K * Nc * 2); };

  unsigned int* pk_in  = pku(256, 256);
  unsigned int* pk_out = pku(256, 256);
  unsigned int* pk_d   = pku(256, 256);
  unsigned int* pk_a   = pku(256, 256);
  unsigned int *pk_sq[3], *pk_sk[3], *pk_sv[3], *pk_so[3], *pk_sf1[3], *pk_sf2[3];
  unsigned int *pk_cq[3], *pk_ck[3], *pk_cv[3], *pk_co[3], *pk_cf1[3], *pk_cf2[3];
  for (int i = 0; i < 3; ++i) {
    pk_sq[i]=pku(256,256); pk_sk[i]=pku(256,256); pk_sv[i]=pku(256,256); pk_so[i]=pku(256,256);
    pk_sf1[i]=pku(256,FFND); pk_sf2[i]=pku(FFND,256);
  }
  for (int i = 0; i < 3; ++i) {
    pk_cq[i]=pku(256,256); pk_ck[i]=pku(256,256); pk_cv[i]=pku(256,256); pk_co[i]=pku(256,256);
    pk_cf1[i]=pku(256,FFND); pk_cf2[i]=pku(FFND,256);
  }
  float* refs_s = (float*)alloc((size_t)rows * 9 * 4);
  float* refs_t = (float*)alloc((size_t)rows * 9 * 4);
  _Float16* rpe_s = (_Float16*)alloc((size_t)B * N * N * 256 * 2);
  _Float16* rpe_t = (_Float16*)alloc((size_t)B * N * N * 256 * 2);
  float* xs_ = (float*)alloc((size_t)rows * 256 * 4);
  float* xt_ = (float*)alloc((size_t)rows * 256 * 4);
  float* Qb  = (float*)alloc((size_t)rows * 256 * 4);
  float* Kb  = (float*)alloc((size_t)rows * 256 * 4);
  float* Vb  = (float*)alloc((size_t)rows * 256 * 4);
  float* Qrb = (float*)alloc((size_t)rows * 1024 * 4);
  float* attn = (float*)alloc((size_t)B * 4 * N * N * 4);
  float* avb = (float*)alloc((size_t)rows * 256 * 4);
  float* yb  = (float*)alloc((size_t)rows * 256 * 4);
  float* hb  = (float*)alloc((size_t)rows * FFND * 4);

  // -------- launch helpers --------
  auto pack = [&](const float* W, unsigned int* P, int K, int Nc) {
    long total = (long)K * Nc / 2;
    k_pack<<<dim3((unsigned)((total + 255) / 256)), 256, 0, stream>>>(W, P, K, Nc);
  };
  auto gemm = [&](const float* A, const unsigned int* P, const float* bias,
                  float* C, int M, int Nc, int K, int relu) {
    k_gemm<<<dim3(M / 16, Nc / 64), 32, 0, stream>>>(A, P, bias, C, M, Nc, K, relu);
  };

  // -------- pack all weights to WMMA f16 fragments --------
  pack(inW, pk_in, 256, 256);  pack(outW, pk_out, 256, 256);
  pack(projD, pk_d, 256, 256); pack(projA, pk_a, 256, 256);
  for (int i = 0; i < 3; ++i) {
    pack(S[i].qw, pk_sq[i], 256, 256); pack(S[i].kw, pk_sk[i], 256, 256);
    pack(S[i].vw, pk_sv[i], 256, 256); pack(S[i].ow, pk_so[i], 256, 256);
    pack(S[i].f1w, pk_sf1[i], 256, FFND); pack(S[i].f2w, pk_sf2[i], FFND, 256);
    pack(Cp[i].qw, pk_cq[i], 256, 256); pack(Cp[i].kw, pk_ck[i], 256, 256);
    pack(Cp[i].vw, pk_cv[i], 256, 256); pack(Cp[i].ow, pk_co[i], 256, 256);
    pack(Cp[i].f1w, pk_cf1[i], 256, FFND); pack(Cp[i].f2w, pk_cf2[i], FFND, 256);
  }

  // -------- geometric structure embedding (once, f16) --------
  k_knn<<<(rows + 255) / 256, 256, 0, stream>>>(src_points, refs_s, B, N);
  k_knn<<<(rows + 255) / 256, 256, 0, stream>>>(tgt_points, refs_t, B, N);
  unsigned gseBlocks = (unsigned)((long)B * N * N / 16);
  k_gse<<<gseBlocks, 128, 0, stream>>>(src_points, refs_s, pk_d, pk_a, rpe_s, N);
  k_gse<<<gseBlocks, 128, 0, stream>>>(tgt_points, refs_t, pk_d, pk_a, rpe_t, N);

  // -------- input projection --------
  gemm(src_feats, pk_in, inB, xs_, rows, 256, 256, 0);
  gemm(tgt_feats, pk_in, inB, xt_, rows, 256, 256, 0);

  auto self_side = [&](float* x, const _Float16* rpe, int i) {
    gemm(x, pk_sq[i], S[i].qb, Qb, rows, 256, 256, 0);
    gemm(x, pk_sk[i], S[i].kb, Kb, rows, 256, 256, 0);
    gemm(x, pk_sv[i], S[i].vb, Vb, rows, 256, 256, 0);
    k_qr<<<rows, 256, 0, stream>>>(Qb, S[i].rpew, Qrb);
    k_scores<<<dim3(N, B), 320, 0, stream>>>(Qb, Kb, Qrb, rpe, attn, N, 1);
    k_av<<<dim3(N, B), 256, 0, stream>>>(attn, Vb, avb, N);
    gemm(avb, pk_so[i], S[i].ob, yb, rows, 256, 256, 0);
    k_add_ln<<<rows, 256, 0, stream>>>(x, yb, S[i].l1g, S[i].l1b, x);
    gemm(x, pk_sf1[i], S[i].f1b, hb, rows, FFND, 256, 1);
    gemm(hb, pk_sf2[i], S[i].f2b, yb, rows, 256, FFND, 0);
    k_add_ln<<<rows, 256, 0, stream>>>(x, yb, S[i].l2g, S[i].l2b, x);
  };
  auto cross_side = [&](float* xq, const float* xkv, int i) {
    gemm(xq, pk_cq[i], Cp[i].qb, Qb, rows, 256, 256, 0);
    gemm(xkv, pk_ck[i], Cp[i].kb, Kb, rows, 256, 256, 0);
    gemm(xkv, pk_cv[i], Cp[i].vb, Vb, rows, 256, 256, 0);
    k_scores<<<dim3(N, B), 320, 0, stream>>>(Qb, Kb, nullptr, nullptr, attn, N, 0);
    k_av<<<dim3(N, B), 256, 0, stream>>>(attn, Vb, avb, N);
    gemm(avb, pk_co[i], Cp[i].ob, yb, rows, 256, 256, 0);
    k_add_ln<<<rows, 256, 0, stream>>>(xq, yb, Cp[i].lag, Cp[i].lab, xq);
    gemm(xq, pk_cf1[i], Cp[i].f1b, hb, rows, FFND, 256, 1);
    gemm(hb, pk_cf2[i], Cp[i].f2b, yb, rows, 256, FFND, 0);
    k_add_ln<<<rows, 256, 0, stream>>>(xq, yb, Cp[i].lfg, Cp[i].lfb, xq);
  };

  for (int i = 0; i < 3; ++i) {
    self_side(xs_, rpe_s, i);
    self_side(xt_, rpe_t, i);
    cross_side(xs_, xt_, i);   // src = cross(src, tgt)
    cross_side(xt_, xs_, i);   // tgt = cross(tgt, updated src)
  }

  // -------- output projection straight into d_out (src then tgt) --------
  float* outp = (float*)d_out;
  gemm(xs_, pk_out, outB, outp, rows, 256, 256, 0);
  gemm(xt_, pk_out, outB, outp + (long)rows * 256, rows, 256, 256, 0);
}
